// AttnDecoder_30262339567935
// MI455X (gfx1250) — compile-verified, matrix operand order checked by reading, not run on hardware
//
#include <hip/hip_runtime.h>
#include <hip/hip_bf16.h>

#define B_ 32
#define T_ 128
#define H_ 256
#define V_ 32000

typedef __attribute__((ext_vector_type(16))) _Float16 v16h;
typedef __attribute__((ext_vector_type(8)))  float    v8f;

union FragAB { uint4 u[2]; v16h v; };

__device__ __forceinline__ float dot4(float4 a, float4 b) {
  return a.x * b.x + a.y * b.y + a.z * b.z + a.w * b.w;
}
__device__ __forceinline__ float sigmoidf_(float x) {
  return 1.f / (1.f + __expf(-x));
}

// ---------------------------------------------------------------------------
// Kernel 1: embedding gather + ReLU  -> enc (B*T, H) f32 in workspace
// ---------------------------------------------------------------------------
__global__ __launch_bounds__(H_) void emb_kernel(const float* __restrict__ emb_table,
                                                 const int* __restrict__ tok,
                                                 float* __restrict__ enc) {
  const int bt = blockIdx.x;      // 0..B*T-1
  const int h  = threadIdx.x;     // 0..H-1
  const int idx = tok[bt];
  float v = emb_table[(size_t)idx * H_ + h];
  enc[(size_t)bt * H_ + h] = v > 0.f ? v : 0.f;
}

// ---------------------------------------------------------------------------
// Kernel 2: lin2_w f32 -> f16 (16.4MB, stays resident in 192MB L2 for GEMM)
// ---------------------------------------------------------------------------
__global__ __launch_bounds__(256) void cvt_kernel(const float* __restrict__ w,
                                                  _Float16* __restrict__ wh, int n) {
  int i = blockIdx.x * 256 + threadIdx.x;
  if (i < n) wh[i] = (_Float16)w[i];
}

// ---------------------------------------------------------------------------
// Kernel 3: full 128-step LSTM + attention scan.
// Recurrence independent per batch row -> 1 WG (1024 thr = 32 waves) per b.
// Thread layout: q = tid>>8 (k-split quarter), r = tid&255 (output index).
// Gates: 1 output per thread. All H-wide GEMVs: 4-way k-split + LDS reduce.
// All weight reads are float4 (global_load_b128), LDS reads float4 (ds b128).
// ---------------------------------------------------------------------------
__global__ __launch_bounds__(1024) void decode_kernel(
    const float* __restrict__ memory, const int* __restrict__ xs_len,
    const float* __restrict__ enc,
    const float* __restrict__ w_ih, const float* __restrict__ w_hh,
    const float* __restrict__ b_ih, const float* __restrict__ b_hh,
    const float* __restrict__ attn_l1, const float* __restrict__ attn_l2,
    const float* __restrict__ lin1_w, const float* __restrict__ lin1_b,
    _Float16* __restrict__ s1_f16,
    float* __restrict__ out_attn, float* __restrict__ out_mem,
    float* __restrict__ out_ctx) {
  __shared__ __align__(16) float sh[H_], sc[H_], sx[H_], shn[H_];
  __shared__ __align__(16) float sxq[H_], sct[H_], sctn[H_];
  __shared__ __align__(16) float sg[4 * H_];
  __shared__ __align__(16) float part[1024];
  __shared__ float satt[T_], red[128];

  const int b = blockIdx.x;
  const int tid = threadIdx.x;
  const int q = tid >> 8;    // 0..3
  const int r = tid & 255;   // 0..255
  const int len = xs_len[b];
  const float* encb = enc + (size_t)b * T_ * H_;

  if (q == 0) {
    sh[r] = memory[b * H_ + r];  // h0 = memory[0, b, :]
    sc[r] = 0.f;                 // ctx0 = 0
  }
  __syncthreads();

  for (int t = 0; t < T_; ++t) {
    if (q == 0) sx[r] = encb[t * H_ + r];
    __syncthreads();

    // ---- gates = [x, ctx] @ w_ih.T + b_ih + h @ w_hh.T + b_hh  (j = tid) ----
    {
      const int j = tid;
      const float4* wi4 = (const float4*)(w_ih + (size_t)j * (2 * H_));
      const float4* wh4 = (const float4*)(w_hh + (size_t)j * H_);
      const float4* sx4 = (const float4*)sx;
      const float4* sc4 = (const float4*)sc;
      const float4* sh4 = (const float4*)sh;
      float acc = b_ih[j] + b_hh[j];
#pragma unroll 8
      for (int k = 0; k < 64; ++k) acc += dot4(wi4[k], sx4[k]);
#pragma unroll 8
      for (int k = 0; k < 64; ++k) acc += dot4(wi4[64 + k], sc4[k]);
#pragma unroll 8
      for (int k = 0; k < 64; ++k) acc += dot4(wh4[k], sh4[k]);
      sg[j] = acc;
    }
    __syncthreads();

    // ---- elementwise LSTM cell (reference uses h as the cell state) ----
    if (q == 0) {
      const float ig = sigmoidf_(sg[r]);
      const float fg = sigmoidf_(sg[H_ + r]);
      const float gg = tanhf(sg[2 * H_ + r]);
      const float og = sigmoidf_(sg[3 * H_ + r]);
      const float cn = fg * sh[r] + ig * gg;
      shn[r] = og * tanhf(cn);
    }
    __syncthreads();

    // ---- xq = h_new @ attn_l1.T  (out r, quarter q covers k=q*64..+64) ----
    {
      const float4* w4 = (const float4*)(attn_l1 + (size_t)r * H_ + q * 64);
      const float4* v4 = (const float4*)(shn + q * 64);
      float acc = 0.f;
#pragma unroll
      for (int k = 0; k < 16; ++k) acc += dot4(w4[k], v4[k]);
      part[tid] = acc;
    }
    __syncthreads();
    if (q == 0) sxq[r] = part[r] + part[256 + r] + part[512 + r] + part[768 + r];
    __syncthreads();

    // ---- att[s] = <enc[b,s,:], xq>  (8 threads per s, k = sub*32..+32) ----
    {
      const int s = tid >> 3, sub = tid & 7;
      const float4* e4 = (const float4*)(encb + s * H_ + sub * 32);
      const float4* x4 = (const float4*)(sxq + sub * 32);
      float acc = 0.f;
#pragma unroll
      for (int k = 0; k < 8; ++k) acc += dot4(e4[k], x4[k]);
      part[tid] = acc;
    }
    __syncthreads();
    if (tid < T_) {
      float a = 0.f;
#pragma unroll
      for (int i = 0; i < 8; ++i) a += part[tid * 8 + i];
      float m = (tid < len) ? a : 0.f;     // seq_mask * att
      if (m == 0.f) m = -1e10f;            // reference quirk: exact-0 masked too
      satt[tid] = m;
    }
    __syncthreads();

    // ---- softmax over S=128 ----
    if (tid < 64) red[tid] = fmaxf(satt[tid], satt[tid + 64]);
    __syncthreads();
    for (int s = 32; s > 0; s >>= 1) {
      if (tid < s) red[tid] = fmaxf(red[tid], red[tid + s]);
      __syncthreads();
    }
    const float mx = red[0];
    __syncthreads();
    if (tid < T_) satt[tid] = __expf(satt[tid] - mx);
    __syncthreads();
    if (tid < 64) red[tid] = satt[tid] + satt[tid + 64];
    __syncthreads();
    for (int s = 32; s > 0; s >>= 1) {
      if (tid < s) red[tid] += red[tid + s];
      __syncthreads();
    }
    const float inv = 1.f / red[0];
    __syncthreads();
    if (tid < T_) {
      satt[tid] *= inv;
      out_attn[((size_t)t * T_ + tid) * B_ + b] = satt[tid];  // (T, S, B)
    }
    __syncthreads();

    // ---- ctx_raw = attn @ enc  (out r, quarter q covers s=q*32..+32) ----
    {
      float acc = 0.f;
#pragma unroll 8
      for (int s = 0; s < 32; ++s) {
        const int ss = q * 32 + s;
        acc += satt[ss] * encb[ss * H_ + r];
      }
      part[tid] = acc;
    }
    __syncthreads();
    if (q == 0) sct[r] = part[r] + part[256 + r] + part[512 + r] + part[768 + r];
    __syncthreads();

    // ---- ctx_new = tanh([ctx_raw, h_new] @ attn_l2.T)  (512-wide k-split) ----
    {
      const float4* w4 = (const float4*)(attn_l2 + (size_t)r * (2 * H_) + q * 128);
      const float4* v4 = (q < 2) ? (const float4*)(sct + q * 128)
                                 : (const float4*)(shn + (q - 2) * 128);
      float acc = 0.f;
#pragma unroll
      for (int k = 0; k < 32; ++k) acc += dot4(w4[k], v4[k]);
      part[tid] = acc;
    }
    __syncthreads();
    if (q == 0)
      sctn[r] = tanhf(part[r] + part[256 + r] + part[512 + r] + part[768 + r]);
    __syncthreads();

    // ---- s1 = tanh(ctx_new @ lin1_w.T + b) -> f16 GEMM A row (b*T + t) ----
    {
      const float4* w4 = (const float4*)(lin1_w + (size_t)r * H_ + q * 64);
      const float4* v4 = (const float4*)(sctn + q * 64);
      float acc = 0.f;
#pragma unroll
      for (int k = 0; k < 16; ++k) acc += dot4(w4[k], v4[k]);
      part[tid] = acc;
    }
    __syncthreads();
    if (q == 0) {
      const float s1 =
          tanhf(lin1_b[r] + part[r] + part[256 + r] + part[512 + r] + part[768 + r]);
      s1_f16[((size_t)b * T_ + t) * H_ + r] = (_Float16)s1;
      sh[r] = shn[r];
      sc[r] = sctn[r];
    }
    __syncthreads();
  }

  if (q == 0) {
    out_mem[b * H_ + r] = sh[r];  // memory_out = hT
    out_ctx[b * H_ + r] = sc[r];  // ctxT
  }
}

// ---------------------------------------------------------------------------
// Kernel 4: logits = s1(4096x256,f16) @ lin2_w^T(256x32000,f16) + bias, via
// v_wmma_f32_16x16x32_f16. 8 waves/WG, 2 N-tiles per wave (A frags reused).
// Fragment layout (ISA 7.12.2): lane l holds row (l&15), K = base..base+7 and
// 16+base..16+base+7 with base = (l<16?0:8) -> two b128 loads per fragment.
// ---------------------------------------------------------------------------
__global__ __launch_bounds__(256) void gemm_kernel(const _Float16* __restrict__ A,
                                                   const _Float16* __restrict__ W,
                                                   const float* __restrict__ bias,
                                                   float* __restrict__ out) {
  const int lane = threadIdx.x & 31;
  const int wave = threadIdx.x >> 5;
  const int m0 = blockIdx.y * 16;
  const int n0 = (blockIdx.x * 16 + wave * 2) * 16;
  const int lmod = lane & 15;
  const int khalf = lane >> 4;

  // A fragments for all 8 K-chunks (K=256 total), kept in VGPRs.
  const _Float16* arow = A + (size_t)(m0 + lmod) * H_ + khalf * 8;
  FragAB a[8];
#pragma unroll
  for (int kc = 0; kc < 8; ++kc) {
    a[kc].u[0] = *(const uint4*)(arow + kc * 32);
    a[kc].u[1] = *(const uint4*)(arow + kc * 32 + 16);
  }

  const int nc0 = n0 + lmod;
  const int nc1 = n0 + 16 + lmod;
  const _Float16* w0 = W + (size_t)nc0 * H_ + khalf * 8;
  const _Float16* w1 = W + (size_t)nc1 * H_ + khalf * 8;

  // Fold bias into initial accumulator (C lane l covers column l&15, all rows)
  const float bv0 = bias[nc0];
  const float bv1 = bias[nc1];
  v8f c0, c1;
#pragma unroll
  for (int r = 0; r < 8; ++r) { c0[r] = bv0; c1[r] = bv1; }

#pragma unroll
  for (int kc = 0; kc < 8; ++kc) {
    FragAB f0, f1;
    f0.u[0] = *(const uint4*)(w0 + kc * 32);
    f0.u[1] = *(const uint4*)(w0 + kc * 32 + 16);
    f1.u[0] = *(const uint4*)(w1 + kc * 32);
    f1.u[1] = *(const uint4*)(w1 + kc * 32 + 16);
    c0 = __builtin_amdgcn_wmma_f32_16x16x32_f16(false, a[kc].v, false, f0.v,
                                                (short)0, c0, false, false);
    c1 = __builtin_amdgcn_wmma_f32_16x16x32_f16(false, a[kc].v, false, f1.v,
                                                (short)0, c1, false, false);
  }

  // D layout: lane l, vgpr r -> (m = r + (l>=16?8:0), n = l&15)
  const int rbase = m0 + (khalf ? 8 : 0);
#pragma unroll
  for (int r = 0; r < 8; ++r) {
    out[(size_t)(rbase + r) * V_ + nc0] = c0[r];
    out[(size_t)(rbase + r) * V_ + nc1] = c1[r];
  }
}

// ---------------------------------------------------------------------------
// Kernel 5: in-place log_softmax over V per row (4096 rows; rows L2-hot)
// ---------------------------------------------------------------------------
__global__ __launch_bounds__(256) void lsm_kernel(float* __restrict__ scores) {
  __shared__ float red[256];
  const int tid = threadIdx.x;
  float* row = scores + (size_t)blockIdx.x * V_;

  float mx = -3.4e38f;
  for (int i = tid; i < V_; i += 256) mx = fmaxf(mx, row[i]);
  red[tid] = mx;
  __syncthreads();
  for (int s = 128; s > 0; s >>= 1) {
    if (tid < s) red[tid] = fmaxf(red[tid], red[tid + s]);
    __syncthreads();
  }
  mx = red[0];
  __syncthreads();

  float sum = 0.f;
  for (int i = tid; i < V_; i += 256) sum += __expf(row[i] - mx);
  red[tid] = sum;
  __syncthreads();
  for (int s = 128; s > 0; s >>= 1) {
    if (tid < s) red[tid] += red[tid + s];
    __syncthreads();
  }
  const float lse = mx + __logf(red[0]);
  __syncthreads();

  for (int i = tid; i < V_; i += 256) row[i] = row[i] - lse;
}

// ---------------------------------------------------------------------------
extern "C" void kernel_launch(void* const* d_in, const int* in_sizes, int n_in,
                              void* d_out, int out_size, void* d_ws, size_t ws_size,
                              hipStream_t stream) {
  (void)in_sizes; (void)n_in; (void)out_size; (void)ws_size;

  const float* memory    = (const float*)d_in[0];
  const int*   tok       = (const int*)d_in[1];
  const int*   xs_len    = (const int*)d_in[2];
  const float* emb_table = (const float*)d_in[3];
  const float* w_ih      = (const float*)d_in[4];
  const float* w_hh      = (const float*)d_in[5];
  const float* b_ih      = (const float*)d_in[6];
  const float* b_hh      = (const float*)d_in[7];
  const float* attn_l1   = (const float*)d_in[8];
  const float* attn_l2   = (const float*)d_in[9];
  const float* lin1_w    = (const float*)d_in[10];
  const float* lin1_b    = (const float*)d_in[11];
  const float* lin2_w    = (const float*)d_in[12];
  const float* lin2_b    = (const float*)d_in[13];

  // d_out: scores(B,T,V) | memory_out(1,B,H) | attn_wts(T,T,B) | ctxT(B,H)
  float* out_scores = (float*)d_out;
  float* out_mem    = out_scores + (size_t)B_ * T_ * V_;
  float* out_attn   = out_mem + (size_t)B_ * H_;
  float* out_ctx    = out_attn + (size_t)T_ * T_ * B_;

  // workspace: enc f32 (4MB) | s1 f16 (2MB) | lin2_w f16 (16.4MB)
  char* ws = (char*)d_ws;
  float*    enc  = (float*)ws;
  _Float16* s1f  = (_Float16*)(ws + (size_t)B_ * T_ * H_ * 4);
  _Float16* w2f  = (_Float16*)(ws + (size_t)B_ * T_ * H_ * 4 + (size_t)B_ * T_ * H_ * 2);

  emb_kernel<<<B_ * T_, H_, 0, stream>>>(emb_table, tok, enc);
  cvt_kernel<<<(V_ * H_ + 255) / 256, 256, 0, stream>>>(lin2_w, w2f, V_ * H_);
  decode_kernel<<<B_, 1024, 0, stream>>>(memory, xs_len, enc, w_ih, w_hh, b_ih, b_hh,
                                         attn_l1, attn_l2, lin1_w, lin1_b,
                                         s1f, out_attn, out_mem, out_ctx);
  gemm_kernel<<<dim3(V_ / 256, (B_ * T_) / 16), 256, 0, stream>>>(s1f, w2f, lin2_b,
                                                                  out_scores);
  lsm_kernel<<<B_ * T_, 256, 0, stream>>>(out_scores);
}
